// LogicLayer_51805895524382
// MI455X (gfx1250) — matrix operand order repaired and checked by reading, not compiled, not run on hardware
//
#include <hip/hip_runtime.h>
#include <hip/hip_bf16.h>

typedef __attribute__((ext_vector_type(16))) _Float16 v16h;
typedef __attribute__((ext_vector_type(8)))  float    v8f;
typedef int v4i __attribute__((vector_size(16)));   // matches builtin's int4

#define AS1 __attribute__((address_space(1)))
#define AS3 __attribute__((address_space(3)))

#define BATCH        4096
#define IN_DIM       16384
#define OUT_DIM      16384
#define ROWS_PER_WG  2
#define THREADS      256

// Gate coefficient matrix: gate g -> [c0, c1, c2, c3]
__constant__ float GATE_M_c[16][4] = {
    {0.f,  0.f,  0.f,  0.f}, {0.f,  0.f,  0.f,  1.f},
    {0.f,  1.f,  0.f, -1.f}, {0.f,  1.f,  0.f,  0.f},
    {0.f,  0.f,  1.f, -1.f}, {0.f,  0.f,  1.f,  0.f},
    {0.f,  1.f,  1.f, -2.f}, {0.f,  1.f,  1.f, -1.f},
    {1.f, -1.f, -1.f,  1.f}, {1.f, -1.f, -1.f,  2.f},
    {1.f,  0.f, -1.f,  0.f}, {1.f,  0.f, -1.f,  1.f},
    {1.f, -1.f,  0.f,  0.f}, {1.f, -1.f,  0.f,  1.f},
    {1.f,  0.f,  0.f, -1.f}, {1.f,  0.f,  0.f,  0.f}};

// -----------------------------------------------------------------------------
// Async global->LDS 128-bit copy helper (gfx1250 GLOBAL_LOAD_ASYNC_TO_LDS_B128,
// tracked by ASYNCcnt). Falls back to a VGPR round-trip copy if the builtin is
// unavailable on this toolchain.
// -----------------------------------------------------------------------------
__device__ __forceinline__ void async_copy_b128(const float4* gsrc, float4* ldst) {
#if defined(__AMDGCN__) && __has_builtin(__builtin_amdgcn_global_load_async_to_lds_b128)
  __builtin_amdgcn_global_load_async_to_lds_b128(
      (AS1 v4i*)gsrc, (AS3 v4i*)ldst, /*offset=*/0, /*cpol=*/0);
#else
  *ldst = *gsrc;
#endif
}

__device__ __forceinline__ void async_copy_wait() {
#if defined(__AMDGCN__) && __has_builtin(__builtin_amdgcn_global_load_async_to_lds_b128)
#if __has_builtin(__builtin_amdgcn_s_wait_asynccnt)
  __builtin_amdgcn_s_wait_asynccnt(0);
#else
  asm volatile("s_wait_asynccnt 0" ::: "memory");
#endif
#endif
}

// -----------------------------------------------------------------------------
// Kernel 1: fold softmax(weights[j, 0:16]) @ GATE_M[16x4] -> coef[j] (float4)
// One wave handles 16 neurons with a single v_wmma_f32_16x16x32_f16:
//   A (16x32 f16) = softmax weights (K=0..15), zero-padded K=16..31
//   B (32x16 f16) = GATE_M in columns N=0..3, zero elsewhere
//   D (16x16 f32) columns 0..3 = the 4 coefficients per neuron.
// -----------------------------------------------------------------------------
__global__ __launch_bounds__(THREADS) void coeff_wmma_kernel(
    const float* __restrict__ w, float4* __restrict__ coef) {
  const int tid  = blockIdx.x * blockDim.x + threadIdx.x;
  const int lane = threadIdx.x & 31;
  const int wave = tid >> 5;
  const int base = wave * 16;      // first neuron of this wave's 16-group
  const int m    = lane & 15;      // row (neuron) / column (coeff) id
  const int hi   = lane >> 4;      // 0: K 0..15 (gates), 1: K 16..31 (zero)

  // Softmax over the 16 gate logits of neuron (base+m). Two lanes per row
  // redundantly compute it (cached reads), which is cheapest at this scale.
  const float* wr = w + (size_t)(base + m) * 16;
  float v[16];
  float mx = wr[0];
#pragma unroll
  for (int i = 0; i < 16; ++i) { v[i] = wr[i]; mx = fmaxf(mx, v[i]); }
  float s = 0.f;
#pragma unroll
  for (int i = 0; i < 16; ++i) { v[i] = __expf(v[i] - mx); s += v[i]; }
  const float inv = 1.0f / s;

  v16h A, B;
#pragma unroll
  for (int i = 0; i < 16; ++i) { A[i] = (_Float16)0.f; B[i] = (_Float16)0.f; }
#pragma unroll
  for (int i = 0; i < 16; ++i) A[i] = (_Float16)(hi == 0 ? v[i] * inv : 0.f);

  const int  mc   = m & 3;
  const bool bcol = (hi == 0) && (m < 4);
#pragma unroll
  for (int e = 0; e < 16; ++e) {
    float bv = bcol ? GATE_M_c[e][mc] : 0.0f;   // branch-free select
    B[e] = (_Float16)bv;
  }

  v8f C = {};
  C = __builtin_amdgcn_wmma_f32_16x16x32_f16(
      /*neg_a=*/false, A, /*neg_b=*/false, B,
      /*c_mod=*/(short)0, C, /*reuse_a=*/false, /*reuse_b=*/false);

  // D layout: lane (hi*16+m) holds column N=m; VGPR r = row (hi*8 + r).
  if (m < 4) {
    float* cp = (float*)coef;
#pragma unroll
    for (int r = 0; r < 8; ++r) {
      int row = hi * 8 + r;
      cp[(size_t)(base + row) * 4 + m] = C[r];
    }
  }
}

// -----------------------------------------------------------------------------
// Kernel 2: the bandwidth kernel. One workgroup per ROWS_PER_WG batch rows:
//   1) async-copy the rows of x into LDS (coalesced B128, ASYNCcnt tracked)
//   2) per output j: gather a,b from LDS, fused 4-coef evaluation, coalesced
//      store. idx/coef arrays are L2/WGP$-resident across all 2048 blocks.
// -----------------------------------------------------------------------------
__global__ __launch_bounds__(THREADS) void logic_gather_kernel(
    const float* __restrict__ x, const float4* __restrict__ coef,
    const int* __restrict__ idx_a, const int* __restrict__ idx_b,
    float* __restrict__ out) {
  extern __shared__ float xs[];          // ROWS_PER_WG * IN_DIM floats (128 KB)
  const int row0 = blockIdx.x * ROWS_PER_WG;
  const int t    = threadIdx.x;

#pragma unroll
  for (int r = 0; r < ROWS_PER_WG; ++r) {
    const float4* src  = (const float4*)(x + (size_t)(row0 + r) * IN_DIM);
    float4*       dst4 = (float4*)(xs + r * IN_DIM);
#pragma unroll 4
    for (int k = t; k < IN_DIM / 4; k += THREADS) {
      async_copy_b128(src + k, dst4 + k);
    }
  }
  async_copy_wait();
  __syncthreads();

  for (int j = t; j < OUT_DIM; j += THREADS) {
    const int    ia = idx_a[j];
    const int    ib = idx_b[j];
    const float4 c  = coef[j];
#pragma unroll
    for (int r = 0; r < ROWS_PER_WG; ++r) {
      const float a = xs[r * IN_DIM + ia];   // ds_load_b32 gather
      const float b = xs[r * IN_DIM + ib];
      out[(size_t)(row0 + r) * OUT_DIM + j] =
          c.x + c.y * a + c.z * b + c.w * (a * b);
    }
  }
}

// -----------------------------------------------------------------------------
// inputs: d_in[0]=x f32[4096,16384], d_in[1]=weights f32[16384,16],
//         d_in[2]=idx_a i32[16384], d_in[3]=idx_b i32[16384]
// output: f32[4096,16384]; d_ws holds coef f32x4[16384] (256 KB)
// -----------------------------------------------------------------------------
extern "C" void kernel_launch(void* const* d_in, const int* in_sizes, int n_in,
                              void* d_out, int out_size, void* d_ws,
                              size_t ws_size, hipStream_t stream) {
  const float* x  = (const float*)d_in[0];
  const float* w  = (const float*)d_in[1];
  const int*   ia = (const int*)d_in[2];
  const int*   ib = (const int*)d_in[3];
  float*       out  = (float*)d_out;
  float4*      coef = (float4*)d_ws;

  // 1024 waves total, 8 waves per 256-thread block -> 128 blocks.
  const int waves_per_block = THREADS / 32;
  const int grid1 = (OUT_DIM / 16) / waves_per_block;
  coeff_wmma_kernel<<<grid1, THREADS, 0, stream>>>(w, coef);

  const size_t lds_bytes = (size_t)ROWS_PER_WG * IN_DIM * sizeof(float);
  logic_gather_kernel<<<BATCH / ROWS_PER_WG, THREADS, lds_bytes, stream>>>(
      x, coef, ia, ib, out);
}
